// MixtureOfExperts_46832323395670
// MI455X (gfx1250) — matrix operand order repaired and checked by reading, not compiled, not run on hardware
//
#include <hip/hip_runtime.h>
#include <math.h>
#include <stdint.h>

// ---------------------------------------------------------------------------
// MoE top-2 (B=2,S=2048,D=1024,H=4096,E=8) for gfx1250 (MI455X, wave32, WMMA)
// Round 3: double-buffered K-chunks staged with GLOBAL_LOAD_ASYNC_TO_LDS_B128
// (ASYNCcnt path, inline asm), overlapping the tensor DMA with the
// v_wmma_f32_16x16x32_bf16 inner loop. One barrier per chunk.
// ---------------------------------------------------------------------------

typedef __bf16 bf16_t;
typedef __attribute__((ext_vector_type(16))) __bf16 v16bf;
typedef __attribute__((ext_vector_type(8)))  __bf16 v8bf;
typedef __attribute__((ext_vector_type(8)))  float  v8f;

#define T_TOK 4096
#define DDIM  1024
#define HDIM  4096
#define NEXP  8
#define BK    128          // K-chunk staged in LDS
#define BM    64           // tokens per block
#define BN    64           // output cols per block

// ---------------------------------------------------------------------------
// Conversions / transposes (one pass, sets up contiguous WMMA B columns)
// ---------------------------------------------------------------------------
__global__ void moe_cvt_x(const float* __restrict__ x, bf16_t* __restrict__ xb, int n) {
    int i = blockIdx.x * blockDim.x + threadIdx.x;
    if (i < n) xb[i] = (bf16_t)x[i];
}

// W1 [E][D][H] fp32  ->  w1t [E][H][D] bf16
__global__ void moe_cvt_w1(const float* __restrict__ W1, bf16_t* __restrict__ w1t) {
    size_t i = (size_t)blockIdx.x * blockDim.x + threadIdx.x;
    size_t total = (size_t)NEXP * DDIM * HDIM;
    if (i >= total) return;
    size_t e = i / ((size_t)DDIM * HDIM);
    size_t r = i - e * (size_t)DDIM * HDIM;
    size_t d = r / HDIM;
    size_t h = r - d * HDIM;
    w1t[(e * HDIM + h) * DDIM + d] = (bf16_t)W1[i];
}

// W2 [E][H][D] fp32  ->  w2t [E][D][H] bf16
__global__ void moe_cvt_w2(const float* __restrict__ W2, bf16_t* __restrict__ w2t) {
    size_t i = (size_t)blockIdx.x * blockDim.x + threadIdx.x;
    size_t total = (size_t)NEXP * HDIM * DDIM;
    if (i >= total) return;
    size_t e = i / ((size_t)HDIM * DDIM);
    size_t r = i - e * (size_t)HDIM * DDIM;
    size_t h = r / DDIM;
    size_t d = r - h * DDIM;
    w2t[(e * DDIM + d) * HDIM + h] = (bf16_t)W2[i];
}

// ---------------------------------------------------------------------------
// Gating: logits = x @ Wg, softmax, top-2, renormalize, scatter to expert lists
// ---------------------------------------------------------------------------
__global__ void moe_gate(const float* __restrict__ x, const float* __restrict__ Wg,
                         int* __restrict__ counts, int* __restrict__ tok_ids,
                         float* __restrict__ wts) {
    int lane = threadIdx.x & 31;
    int wave = threadIdx.x >> 5;
    int t = blockIdx.x * (blockDim.x >> 5) + wave;
    if (t >= T_TOK) return;

    float acc[NEXP];
#pragma unroll
    for (int e = 0; e < NEXP; ++e) acc[e] = 0.0f;

    const float* xr = x + (size_t)t * DDIM;
    for (int d = lane; d < DDIM; d += 32) {
        float xv = xr[d];
        const float* wr = Wg + (size_t)d * NEXP;
#pragma unroll
        for (int e = 0; e < NEXP; ++e) acc[e] += xv * wr[e];
    }
#pragma unroll
    for (int e = 0; e < NEXP; ++e)
        for (int off = 16; off > 0; off >>= 1)
            acc[e] += __shfl_down(acc[e], off, 32);

    if (lane == 0) {
        float m = acc[0];
#pragma unroll
        for (int e = 1; e < NEXP; ++e) m = fmaxf(m, acc[e]);
        float p[NEXP];
        float s = 0.0f;
#pragma unroll
        for (int e = 0; e < NEXP; ++e) { p[e] = __expf(acc[e] - m); s += p[e]; }
        float inv = 1.0f / s;
#pragma unroll
        for (int e = 0; e < NEXP; ++e) p[e] *= inv;
        int i0 = 0;
#pragma unroll
        for (int e = 1; e < NEXP; ++e) if (p[e] > p[i0]) i0 = e;
        int i1 = (i0 == 0) ? 1 : 0;
#pragma unroll
        for (int e = 0; e < NEXP; ++e) if (e != i0 && p[e] > p[i1]) i1 = e;
        float w0 = p[i0], w1 = p[i1];
        float sw = 1.0f / (w0 + w1);
        w0 *= sw; w1 *= sw;
        int s0 = atomicAdd(&counts[i0], 1);
        tok_ids[i0 * T_TOK + s0] = t;  wts[i0 * T_TOK + s0] = w0;
        int s1 = atomicAdd(&counts[i1], 1);
        tok_ids[i1 * T_TOK + s1] = t;  wts[i1 * T_TOK + s1] = w1;
    }
}

// ---------------------------------------------------------------------------
// Async global -> LDS copy of 128 bytes per thread (8x b128), ASYNCcnt path.
// LDS address = low 32 bits of the flat address (ISA 10.2 aperture rule).
// ---------------------------------------------------------------------------
__device__ __forceinline__ void async_copy128(const bf16_t* gsrc, bf16_t* ldst) {
#pragma unroll
    for (int q = 0; q < 8; ++q) {
        unsigned    l = (unsigned)(uintptr_t)(ldst + q * 8);
        const void* g = (const void*)(gsrc + q * 8);
        asm volatile("global_load_async_to_lds_b128 %0, %1, off"
                     :: "v"(l), "v"(g) : "memory");
    }
}
__device__ __forceinline__ void wait_async0() {
    asm volatile("s_wait_asynccnt 0" ::: "memory");
}

// ---------------------------------------------------------------------------
// WMMA fragment helpers (16-bit A 16x32 / B 32x16 layouts, ISA 7.12.2)
// ---------------------------------------------------------------------------
__device__ __forceinline__ v16bf load_fragA(const bf16_t* row, int k, int half) {
    v8bf a0 = *(const v8bf*)(row + k + half * 8);
    v8bf a1 = *(const v8bf*)(row + k + 16 + half * 8);
    return __builtin_shufflevector(a0, a1, 0,1,2,3,4,5,6,7,8,9,10,11,12,13,14,15);
}
__device__ __forceinline__ v16bf load_fragB(const bf16_t* col, int k, int half) {
    v8bf b0 = *(const v8bf*)(col + k + half * 8);
    v8bf b1 = *(const v8bf*)(col + k + 16 + half * 8);
    return __builtin_shufflevector(b0, b1, 0,1,2,3,4,5,6,7,8,9,10,11,12,13,14,15);
}

// ---------------------------------------------------------------------------
// GEMM1: h[slot, :H] = gelu( xb[tok] @ W1_e + b1_e )
// Block: 4 waves, 64 tokens x 64 cols, double-buffered async K-chunks.
// ---------------------------------------------------------------------------
__global__ void __launch_bounds__(128)
moe_gemm1(const bf16_t* __restrict__ xb,
          const bf16_t* __restrict__ w1t,     // [E][H][D]
          const float*  __restrict__ b1,      // [E][H]
          const int*    __restrict__ counts,
          const int*    __restrict__ tok_ids, // [E][T]
          bf16_t*       __restrict__ hbuf,    // [T][H]
          int e) {
    __shared__ bf16_t As[2 * BM * BK];
    __shared__ bf16_t Bs[2 * BN * BK];
    __shared__ int    tids[BM];

    int tx   = threadIdx.x;
    int lane = tx & 31;
    int wave = tx >> 5;
    int cblk = blockIdx.x;
    int tblk = blockIdx.y;
    int cnt  = counts[e];
    if (tblk * BM >= cnt) return;   // block-uniform

    if (tx < BM) tids[tx] = tok_ids[e * T_TOK + tblk * BM + tx];
    __syncthreads();

    int r    = lane & 15;
    int half = lane >> 4;
    int srow  = tx >> 1;            // staging: 2 threads per row/col
    int spart = (tx & 1) * (BK / 2);

    const bf16_t* abase = xb + (size_t)tids[srow] * DDIM + spart;
    const bf16_t* bbase = w1t + ((size_t)e * HDIM + (size_t)cblk * BN + srow) * DDIM + spart;
    bf16_t* asd = As + srow * BK + spart;
    bf16_t* bsd = Bs + srow * BK + spart;

    // prologue: chunk 0 into buffer 0
    async_copy128(abase, asd);
    async_copy128(bbase, bsd);

    v8f acc[4] = {};
    int nchunk = DDIM / BK;
    for (int i = 0; i < nchunk; ++i) {
        int p = i & 1;
        wait_async0();              // my chunk-i copies into buffer p done
        __syncthreads();            // everyone's done (and all have left chunk i-1)
        if (i + 1 < nchunk) {       // overlap: DMA chunk i+1 into buffer p^1
            int k1 = (i + 1) * BK;
            int q1 = (p ^ 1) ? BM * BK : 0;   // (p^1)*BM*BK
            async_copy128(abase + k1, asd + q1);
            async_copy128(bbase + k1, bsd + ((p ^ 1) ? BN * BK : 0));
        }
        const bf16_t* arow = As + p * (BM * BK) + (wave * 16 + r) * BK;
        const bf16_t* bcb  = Bs + p * (BN * BK);
#pragma unroll
        for (int kk = 0; kk < BK; kk += 32) {
            v16bf Af = load_fragA(arow, kk, half);
#pragma unroll
            for (int c = 0; c < 4; ++c) {
                v16bf Bf = load_fragB(bcb + (c * 16 + r) * BK, kk, half);
                acc[c] = __builtin_amdgcn_wmma_f32_16x16x32_bf16(
                    false, Af, false, Bf, (short)0, acc[c], false, false);
            }
        }
    }

    // epilogue: bias + exact gelu, store bf16 h
#pragma unroll
    for (int c = 0; c < 4; ++c) {
        int col  = cblk * BN + c * 16 + r;
        float bias = b1[(size_t)e * HDIM + col];
#pragma unroll
        for (int i = 0; i < 8; ++i) {
            int row = tblk * BM + wave * 16 + half * 8 + i;
            float v = acc[c][i] + bias;
            v = 0.5f * v * (1.0f + erff(v * 0.70710678118654752f));
            hbuf[(size_t)row * HDIM + col] = (bf16_t)v;
        }
    }
}

// ---------------------------------------------------------------------------
// GEMM2: out[tok, :D] += w_tok * ( h[slot] @ W2_e + b2_e )
// ---------------------------------------------------------------------------
__global__ void __launch_bounds__(128)
moe_gemm2(const bf16_t* __restrict__ hbuf,    // [T][H]
          const bf16_t* __restrict__ w2t,     // [E][D][H]
          const float*  __restrict__ b2,      // [E][D]
          const int*    __restrict__ counts,
          const int*    __restrict__ tok_ids,
          const float*  __restrict__ wts,
          float*        __restrict__ out,     // [T][D]
          int e) {
    __shared__ bf16_t As[2 * BM * BK];
    __shared__ bf16_t Bs[2 * BN * BK];

    int tx   = threadIdx.x;
    int lane = tx & 31;
    int wave = tx >> 5;
    int cblk = blockIdx.x;
    int tblk = blockIdx.y;
    int cnt  = counts[e];
    if (tblk * BM >= cnt) return;

    int r    = lane & 15;
    int half = lane >> 4;
    int srow  = tx >> 1;
    int spart = (tx & 1) * (BK / 2);

    const bf16_t* abase = hbuf + (size_t)(tblk * BM + srow) * HDIM + spart;
    const bf16_t* bbase = w2t + ((size_t)e * DDIM + (size_t)cblk * BN + srow) * HDIM + spart;
    bf16_t* asd = As + srow * BK + spart;
    bf16_t* bsd = Bs + srow * BK + spart;

    async_copy128(abase, asd);
    async_copy128(bbase, bsd);

    v8f acc[4] = {};
    int nchunk = HDIM / BK;
    for (int i = 0; i < nchunk; ++i) {
        int p = i & 1;
        wait_async0();
        __syncthreads();
        if (i + 1 < nchunk) {
            int k1 = (i + 1) * BK;
            async_copy128(abase + k1, asd + ((p ^ 1) ? BM * BK : 0));
            async_copy128(bbase + k1, bsd + ((p ^ 1) ? BN * BK : 0));
        }
        const bf16_t* arow = As + p * (BM * BK) + (wave * 16 + r) * BK;
        const bf16_t* bcb  = Bs + p * (BN * BK);
#pragma unroll
        for (int kk = 0; kk < BK; kk += 32) {
            v16bf Af = load_fragA(arow, kk, half);
#pragma unroll
            for (int c = 0; c < 4; ++c) {
                v16bf Bf = load_fragB(bcb + (c * 16 + r) * BK, kk, half);
                acc[c] = __builtin_amdgcn_wmma_f32_16x16x32_bf16(
                    false, Af, false, Bf, (short)0, acc[c], false, false);
            }
        }
    }

    // epilogue: bias, token weight, atomic combine into out
#pragma unroll
    for (int c = 0; c < 4; ++c) {
        int col  = cblk * BN + c * 16 + r;
        float bias = b2[(size_t)e * DDIM + col];
#pragma unroll
        for (int i = 0; i < 8; ++i) {
            int slot = e * T_TOK + tblk * BM + wave * 16 + half * 8 + i;
            int tok  = tok_ids[slot];
            float w  = wts[slot];        // 0 for padding rows
            unsafeAtomicAdd(&out[(size_t)tok * DDIM + col], (acc[c][i] + bias) * w);
        }
    }
}

// ---------------------------------------------------------------------------
// Host launcher (graph-capture safe: only async ops on `stream`)
// ---------------------------------------------------------------------------
extern "C" void kernel_launch(void* const* d_in, const int* in_sizes, int n_in,
                              void* d_out, int out_size, void* d_ws, size_t ws_size,
                              hipStream_t stream) {
    const float* x  = (const float*)d_in[0];
    const float* Wg = (const float*)d_in[1];
    const float* W1 = (const float*)d_in[2];
    const float* b1 = (const float*)d_in[3];
    const float* W2 = (const float*)d_in[4];
    const float* b2 = (const float*)d_in[5];
    float* out = (float*)d_out;

    char* ws = (char*)d_ws;
    size_t off = 0;
    int*    counts  = (int*)(ws + off);    off += 256;
    int*    tok_ids = (int*)(ws + off);    off += (size_t)NEXP * T_TOK * 4;
    float*  wts     = (float*)(ws + off);  off += (size_t)NEXP * T_TOK * 4;
    bf16_t* xb      = (bf16_t*)(ws + off); off += (size_t)T_TOK * DDIM * 2;
    bf16_t* w1t     = (bf16_t*)(ws + off); off += (size_t)NEXP * DDIM * HDIM * 2;
    bf16_t* w2t     = (bf16_t*)(ws + off); off += (size_t)NEXP * HDIM * DDIM * 2;
    bf16_t* hbuf    = (bf16_t*)(ws + off); off += (size_t)T_TOK * HDIM * 2;

    hipMemsetAsync(out, 0, (size_t)out_size * sizeof(float), stream);
    hipMemsetAsync(counts, 0, 256 + (size_t)NEXP * T_TOK * 8, stream);

    {
        int n = T_TOK * DDIM;
        moe_cvt_x<<<(n + 255) / 256, 256, 0, stream>>>(x, xb, n);
        size_t tw = (size_t)NEXP * DDIM * HDIM;
        moe_cvt_w1<<<(unsigned)((tw + 255) / 256), 256, 0, stream>>>(W1, w1t);
        moe_cvt_w2<<<(unsigned)((tw + 255) / 256), 256, 0, stream>>>(W2, w2t);
    }

    moe_gate<<<T_TOK / 8, 256, 0, stream>>>(x, Wg, counts, tok_ids, wts);

    for (int e = 0; e < NEXP; ++e) {
        moe_gemm1<<<dim3(HDIM / BN, T_TOK / BM), 128, 0, stream>>>(
            xb, w1t, b1, counts, tok_ids, hbuf, e);
        moe_gemm2<<<dim3(DDIM / BN, T_TOK / BM), 128, 0, stream>>>(
            hbuf, w2t, b2, counts, tok_ids, wts, out, e);
    }
}
